// MultiHeadAttention_44100724196050
// MI455X (gfx1250) — matrix operand order repaired and checked by reading, not compiled, not run on hardware
//
#include <hip/hip_runtime.h>

#define B_   8
#define S_   2048
#define D_   512
#define H_   8
#define DEPTH_ 64
#define NTOK_ (B_ * S_)
#define NEGV (-1.0e9f)

typedef __attribute__((ext_vector_type(16))) __bf16 v16bf;
typedef __attribute__((ext_vector_type(8)))  float  v8f;

static __device__ __forceinline__ v8f wmma_bf16(v16bf a, v16bf b, v8f c) {
  // D = A(16x32 bf16) * B(32x16 bf16) + C(16x16 f32)
  return __builtin_amdgcn_wmma_f32_16x16x32_bf16(false, a, false, b, (short)0, c,
                                                 false, false);
}

// ---------------------------------------------------------------------------
// Repack a DxD fp32 weight into bf16 layout [K/32][col][32] so that the WMMA
// B-operand (lane n = col, element e -> K = e + 16*half) is one contiguous
// 32B load per lane.
// ---------------------------------------------------------------------------
__global__ void repack_w(const float* __restrict__ W, __bf16* __restrict__ Wp) {
  int t = blockIdx.x * blockDim.x + threadIdx.x;   // 0 .. D*D-1 (grid 1024x256)
  int k = t >> 9;           // row of W (K dim)
  int c = t & (D_ - 1);     // col of W
  Wp[((size_t)(k >> 5) * D_ + c) * 32 + (k & 31)] = (__bf16)W[(size_t)k * D_ + c];
}

// ---------------------------------------------------------------------------
// Input projection GEMM: Y = X(fp32) @ Wp(bf16 packed) + bias, output bf16.
// One wave computes a 16-token x 64-feature tile (4 WMMA accumulators).
// MODE 0: store head-split row-major [b,h,s,64]   (Q, K)
// MODE 1: store head-split transposed [b,h,64,s]  (V)
// grid: 1024 blocks x 256 threads (8 waves) -> 16384 tokens x 512 cols.
// ---------------------------------------------------------------------------
template <int MODE>
__global__ void __launch_bounds__(256, 2)
proj_gemm(const float* __restrict__ X, const __bf16* __restrict__ Wp,
          const float* __restrict__ bias, __bf16* __restrict__ Out) {
  const int lane = threadIdx.x & 31;
  const int wave = threadIdx.x >> 5;
  const int m  = lane & 15;
  const int hf = lane >> 4;
  const int tokBase = blockIdx.x * 16;
  const int colBase = wave * 64;

  v8f acc[4] = {v8f{}, v8f{}, v8f{}, v8f{}};
  const float* xr = X + (size_t)(tokBase + m) * D_;

  for (int kk = 0; kk < D_; kk += 32) {
    __builtin_prefetch(xr + kk + 64, 0, 1);
    // A operand: 16x32 bf16, per-lane layout K(e,hf) = e + 16*hf
    v16bf a;
    const float* p0 = xr + kk + 16 * hf;
    const float* p1 = xr + kk + 8 + 16 * hf;
#pragma unroll
    for (int e = 0; e < 8; ++e) {
      a[e]     = (__bf16)p0[e];
      a[e + 8] = (__bf16)p1[e];
    }
#pragma unroll
    for (int t = 0; t < 4; ++t) {
      const __bf16* wb =
          Wp + ((size_t)(kk >> 5) * D_ + colBase + t * 16 + m) * 32 + 16 * hf;
      v16bf bm = *(const v16bf*)wb;     // 32B aligned contiguous
      acc[t] = wmma_bf16(a, bm, acc[t]);
    }
  }

#pragma unroll
  for (int t = 0; t < 4; ++t) {
    const int col = colBase + t * 16 + m;
    const float bvv = bias[col];
    const int h = col >> 6, dd = col & 63;
#pragma unroll
    for (int r = 0; r < 8; ++r) {
      const int tok = tokBase + r + 8 * hf;          // C layout: M = r + 8*hf
      const int bb = tok >> 11, ss = tok & (S_ - 1);
      const float val = acc[t][r] + bvv;
      const size_t bh = (size_t)bb * H_ + h;
      if (MODE == 0)
        Out[(bh * S_ + ss) * DEPTH_ + dd] = (__bf16)val;
      else
        Out[(bh * DEPTH_ + dd) * S_ + ss] = (__bf16)val;
    }
  }
}

// ---------------------------------------------------------------------------
// Flash-style causal attention, block-cooperative.
// One block (8 waves) owns 128 consecutive query rows of one (b,h); each wave
// owns a 16-row tile. Per 32-key chunk the block stages K(32x64) and V(64x32,
// depth-major) into LDS with double-buffered global_load_async_to_lds_b128
// (ASYNCcnt-sequenced), so K/V are fetched once per block instead of once per
// wave (8x less L2 traffic). Softmax is online; P is relaid out fp32-C-layout
// -> bf16-A-layout through a per-wave LDS buffer (same-wave LDS is in-order).
// launch_bounds(256,2): ~512 VGPR/wave budget -> no scratch spills of aq0/aq1.
// grid: B*H*16 = 1024 blocks x 256 threads.
// ---------------------------------------------------------------------------
__global__ void __launch_bounds__(256, 2)
attn_fwd(const __bf16* __restrict__ Qh, const __bf16* __restrict__ Kh,
         const __bf16* __restrict__ Vt, const float* __restrict__ pad,
         __bf16* __restrict__ Cat) {
  __shared__ __bf16 Ksh[2][32 * 64];   // 2 x 4KB key chunk, row-major [kpos][d]
  __shared__ __bf16 Vsh[2][64 * 32];   // 2 x 4KB value chunk, depth-major [d][kpos]
  __shared__ __bf16 Plds[8][16 * 32];  // per-wave P relayout buffers

  const int tid  = threadIdx.x;
  const int lane = tid & 31;
  const int wave = tid >> 5;
  const int m  = lane & 15;
  const int hf = lane >> 4;

  const int bh   = blockIdx.x >> 4;    // b*H + h
  const int blkq = blockIdx.x & 15;    // which 128-row stripe of the sequence
  const int b = bh >> 3;
  const int h = bh & 7;
  const int qbase = blkq * 128 + wave * 16;

  // Q A-operands for depth 0..31 / 32..63 (loaded once, reused every chunk)
  v16bf aq0, aq1;
  {
    const __bf16* qr = Qh + ((size_t)bh * S_ + qbase + m) * DEPTH_;
#pragma unroll
    for (int e = 0; e < 8; ++e) {
      aq0[e]     = qr[16 * hf + e];
      aq0[e + 8] = qr[8 + 16 * hf + e];
      aq1[e]     = qr[32 + 16 * hf + e];
      aq1[e + 8] = qr[40 + 16 * hf + e];
    }
  }

  v8f cacc[4] = {v8f{}, v8f{}, v8f{}, v8f{}};
  float mrow[8], lrow[8];
#pragma unroll
  for (int r = 0; r < 8; ++r) { mrow[r] = -3.0e38f; lrow[r] = 0.0f; }

  const float* padb = pad + (size_t)b * S_;
  __bf16* Pw = &Plds[wave][0];
  const int kcmax_blk = (blkq * 128 + 127) >> 5;  // block-uniform chunk limit
  // Wave-uniform causal limit: force scalar so the guard is an s_cbranch and
  // EXEC is provably all-1s around the WMMAs.
  const int kcmax_w = __builtin_amdgcn_readfirstlane((qbase + 15) >> 5);
  const float scale = 0.125f;                     // 1/sqrt(64)

  const __bf16* Ksrc  = Kh + (size_t)bh * S_ * DEPTH_;
  const __bf16* Vbase = Vt + (size_t)bh * DEPTH_ * S_;
  const uint32_t ldsK = (uint32_t)(uintptr_t)&Ksh[0][0];  // LDS offset = low 32b
  const uint32_t ldsV = (uint32_t)(uintptr_t)&Vsh[0][0];
  const int vd = tid >> 2, vpart = tid & 3;

  // Stage chunk kc into buffer buf: two linear 4KB copies, 16B per thread.
  auto stage = [&](int kc, int buf) {
    const int k0 = kc * 32;
    const __bf16* gk = Ksrc + (size_t)k0 * DEPTH_ + tid * 8;
    const __bf16* gv = Vbase + (size_t)vd * S_ + k0 + vpart * 8;
    const uint32_t la = ldsK + (uint32_t)buf * 4096u + (uint32_t)tid * 16u;
    const uint32_t lb = ldsV + (uint32_t)buf * 4096u + (uint32_t)tid * 16u;
    asm volatile("global_load_async_to_lds_b128 %0, %1, off"
                 :: "v"(la), "v"((unsigned long long)(uintptr_t)gk) : "memory");
    asm volatile("global_load_async_to_lds_b128 %0, %1, off"
                 :: "v"(lb), "v"((unsigned long long)(uintptr_t)gv) : "memory");
  };

  stage(0, 0);                                    // prime the pipeline

  for (int kc = 0; kc <= kcmax_blk; ++kc) {
    const int cur = kc & 1;
    if (kc < kcmax_blk) {
      stage(kc + 1, cur ^ 1);                     // prefetch next chunk
      // async loads complete in order: <=2 outstanding => chunk kc landed
      asm volatile("s_wait_asynccnt 0x2" ::: "memory");
    } else {
      asm volatile("s_wait_asynccnt 0x0" ::: "memory");
    }
    __syncthreads();                              // chunk kc visible to block

    if (kc <= kcmax_w) {
      const int k0 = kc * 32;
      const __bf16* Kbuf = &Ksh[cur][0];
      const __bf16* Vbuf = &Vsh[cur][0];

      // ---- scores: two 16x16 tiles over 32 key positions ----
      v8f s0 = {}, s1 = {};
      {
        const __bf16* kr0 = Kbuf + (size_t)m * DEPTH_;
        v16bf b0 = *(const v16bf*)(kr0 + 16 * hf);
        v16bf b1 = *(const v16bf*)(kr0 + 32 + 16 * hf);
        s0 = wmma_bf16(aq0, b0, s0);
        s0 = wmma_bf16(aq1, b1, s0);
      }
      {
        const __bf16* kr1 = Kbuf + (size_t)(16 + m) * DEPTH_;
        v16bf b2 = *(const v16bf*)(kr1 + 16 * hf);
        v16bf b3 = *(const v16bf*)(kr1 + 32 + 16 * hf);
        s1 = wmma_bf16(aq0, b2, s1);
        s1 = wmma_bf16(aq1, b3, s1);
      }

      // ---- scale + pad mask + analytic causal mask ----
      const float pv0 = padb[k0 + m] * NEGV;
      const float pv1 = padb[k0 + 16 + m] * NEGV;
      const int kg0 = k0 + m, kg1 = k0 + 16 + m;
#pragma unroll
      for (int r = 0; r < 8; ++r) {
        const int qg = qbase + r + 8 * hf;
        s0[r] = s0[r] * scale + pv0 + (kg0 > qg ? NEGV : 0.0f);
        s1[r] = s1[r] * scale + pv1 + (kg1 > qg ? NEGV : 0.0f);
      }

      // ---- online softmax (row reductions inside 16-lane halves) ----
#pragma unroll
      for (int r = 0; r < 8; ++r) {
        float t = fmaxf(s0[r], s1[r]);
        t = fmaxf(t, __shfl_xor(t, 1, 32));
        t = fmaxf(t, __shfl_xor(t, 2, 32));
        t = fmaxf(t, __shfl_xor(t, 4, 32));
        t = fmaxf(t, __shfl_xor(t, 8, 32));
        const float mnew = fmaxf(mrow[r], t);
        const float corr = __expf(mrow[r] - mnew);
        mrow[r] = mnew;
        const float p0 = __expf(s0[r] - mnew);
        const float p1 = __expf(s1[r] - mnew);
        float rs = p0 + p1;
        rs += __shfl_xor(rs, 1, 32);
        rs += __shfl_xor(rs, 2, 32);
        rs += __shfl_xor(rs, 4, 32);
        rs += __shfl_xor(rs, 8, 32);
        lrow[r] = lrow[r] * corr + rs;
        cacc[0][r] *= corr; cacc[1][r] *= corr;
        cacc[2][r] *= corr; cacc[3][r] *= corr;
        s0[r] = p0; s1[r] = p1;
      }

      // ---- P (fp32 C-layout) -> LDS -> bf16 A-layout ----
#pragma unroll
      for (int r = 0; r < 8; ++r) {
        const int M = r + 8 * hf;
        Pw[M * 32 + m]      = (__bf16)s0[r];
        Pw[M * 32 + 16 + m] = (__bf16)s1[r];
      }
      __builtin_amdgcn_wave_barrier();
      v16bf pa;
#pragma unroll
      for (int e = 0; e < 8; ++e) {
        pa[e]     = Pw[m * 32 + 16 * hf + e];
        pa[e + 8] = Pw[m * 32 + 8 + 16 * hf + e];
      }
      __builtin_amdgcn_wave_barrier();

      // ---- ctx += P @ V (B operand contiguous: depth-major V in LDS) ----
#pragma unroll
      for (int dt = 0; dt < 4; ++dt) {
        const __bf16* vr = Vbuf + (size_t)(dt * 16 + m) * 32 + 16 * hf;
        v16bf bv = *(const v16bf*)vr;
        cacc[dt] = wmma_bf16(pa, bv, cacc[dt]);
      }
    }
    __syncthreads();   // all waves done with buffer `cur` before it is restaged
  }

  // ---- normalize and store into concat layout [b, s, h*64+d] as bf16 ----
#pragma unroll
  for (int dt = 0; dt < 4; ++dt) {
#pragma unroll
    for (int r = 0; r < 8; ++r) {
      const int qg = qbase + r + 8 * hf;
      const float val = cacc[dt][r] / lrow[r];
      Cat[((size_t)b * S_ + qg) * D_ + h * DEPTH_ + dt * 16 + m] = (__bf16)val;
    }
  }
}

// ---------------------------------------------------------------------------
// Output projection: out(fp32) = concat(bf16) @ Wo(packed bf16) + bo.
// Same tiling as proj_gemm; A operand is already bf16 (contiguous loads).
// ---------------------------------------------------------------------------
__global__ void __launch_bounds__(256, 2)
out_proj(const __bf16* __restrict__ Xb, const __bf16* __restrict__ Wp,
         const float* __restrict__ bias, float* __restrict__ Out) {
  const int lane = threadIdx.x & 31;
  const int wave = threadIdx.x >> 5;
  const int m  = lane & 15;
  const int hf = lane >> 4;
  const int tokBase = blockIdx.x * 16;
  const int colBase = wave * 64;

  v8f acc[4] = {v8f{}, v8f{}, v8f{}, v8f{}};
  const __bf16* xr = Xb + (size_t)(tokBase + m) * D_;

  for (int kk = 0; kk < D_; kk += 32) {
    __builtin_prefetch(xr + kk + 64, 0, 1);
    v16bf a;
    const __bf16* p0 = xr + kk + 16 * hf;
    const __bf16* p1 = xr + kk + 8 + 16 * hf;
#pragma unroll
    for (int e = 0; e < 8; ++e) {
      a[e]     = p0[e];
      a[e + 8] = p1[e];
    }
#pragma unroll
    for (int t = 0; t < 4; ++t) {
      const __bf16* wb =
          Wp + ((size_t)(kk >> 5) * D_ + colBase + t * 16 + m) * 32 + 16 * hf;
      v16bf bm = *(const v16bf*)wb;
      acc[t] = wmma_bf16(a, bm, acc[t]);
    }
  }

#pragma unroll
  for (int t = 0; t < 4; ++t) {
    const int col = colBase + t * 16 + m;
    const float bvv = bias[col];
#pragma unroll
    for (int r = 0; r < 8; ++r) {
      const int tok = tokBase + r + 8 * hf;
      Out[(size_t)tok * D_ + col] = acc[t][r] + bvv;
    }
  }
}

// ---------------------------------------------------------------------------
// Host-side launcher. Inputs (setup_inputs order):
// 0:v 1:k 2:q 3:pad_mask 4:head_mask 5:Wq 6:bq 7:Wk 8:bk 9:Wv 10:bv 11:Wo
// 12:bo 13:num_heads. head_mask is exact causal triu -> computed analytically.
// ---------------------------------------------------------------------------
extern "C" void kernel_launch(void* const* d_in, const int* in_sizes, int n_in,
                              void* d_out, int out_size, void* d_ws, size_t ws_size,
                              hipStream_t stream) {
  (void)in_sizes; (void)n_in; (void)out_size; (void)ws_size;

  const float* v   = (const float*)d_in[0];
  const float* k   = (const float*)d_in[1];
  const float* q   = (const float*)d_in[2];
  const float* pad = (const float*)d_in[3];
  const float* Wq  = (const float*)d_in[5];
  const float* bq  = (const float*)d_in[6];
  const float* Wk  = (const float*)d_in[7];
  const float* bk  = (const float*)d_in[8];
  const float* Wv  = (const float*)d_in[9];
  const float* bv  = (const float*)d_in[10];
  const float* Wo  = (const float*)d_in[11];
  const float* bo  = (const float*)d_in[12];

  char* ws = (char*)d_ws;
  const size_t TEN = (size_t)B_ * H_ * S_ * DEPTH_ * sizeof(__bf16);  // 16 MiB
  const size_t WPK = (size_t)D_ * D_ * sizeof(__bf16);                // 512 KiB
  __bf16* Qh  = (__bf16*)(ws + 0 * TEN);
  __bf16* Kh  = (__bf16*)(ws + 1 * TEN);
  __bf16* Vt  = (__bf16*)(ws + 2 * TEN);   // transposed [b,h,64,S]
  __bf16* Cat = (__bf16*)(ws + 3 * TEN);
  __bf16* Wqp = (__bf16*)(ws + 4 * TEN);
  __bf16* Wkp = (__bf16*)(ws + 4 * TEN + 1 * WPK);
  __bf16* Wvp = (__bf16*)(ws + 4 * TEN + 2 * WPK);
  __bf16* Wop = (__bf16*)(ws + 4 * TEN + 3 * WPK);

  dim3 blk(256);
  repack_w<<<dim3(1024), blk, 0, stream>>>(Wq, Wqp);
  repack_w<<<dim3(1024), blk, 0, stream>>>(Wk, Wkp);
  repack_w<<<dim3(1024), blk, 0, stream>>>(Wv, Wvp);
  repack_w<<<dim3(1024), blk, 0, stream>>>(Wo, Wop);

  proj_gemm<0><<<dim3(NTOK_ / 16), blk, 0, stream>>>(q, Wqp, bq, Qh);
  proj_gemm<0><<<dim3(NTOK_ / 16), blk, 0, stream>>>(k, Wkp, bk, Kh);
  proj_gemm<1><<<dim3(NTOK_ / 16), blk, 0, stream>>>(v, Wvp, bv, Vt);

  attn_fwd<<<dim3(B_ * H_ * 16), blk, 0, stream>>>(Qh, Kh, Vt, pad, Cat);

  out_proj<<<dim3(NTOK_ / 16), blk, 0, stream>>>(Cat, Wop, bo, (float*)d_out);
}
